// MambaResidual_85478439125216
// MI455X (gfx1250) — compile-verified, matrix operand order checked by reading, not run on hardware
//
#include <hip/hip_runtime.h>

#define DM   64
#define DS   16
#define DI   128
#define RK   4
#define KC   4
#define BB   8
#define LL   4096
#define TOK  (BB * LL)      // 32768 tokens
#define NTIL (TOK / 16)     // 2048 16-token tiles
#define XPW  (RK + 2 * DS)  // 36
#define EPSV 1e-6f

typedef __attribute__((ext_vector_type(16))) _Float16 v16h;
typedef __attribute__((ext_vector_type(8)))  _Float16 v8h;
typedef __attribute__((ext_vector_type(8)))  float    v8f;

// Async global->LDS (gfx1250 GLOBAL_LOAD_ASYNC_TO_LDS_B128), guarded so a
// toolchain without the builtin still compiles via the fallback path.
#if defined(__has_builtin)
#if __has_builtin(__builtin_amdgcn_global_load_async_to_lds_b128) && \
    __has_builtin(__builtin_amdgcn_s_wait_asynccnt)
#define USE_ASYNC_LDS 1
#endif
#endif

#if defined(USE_ASYNC_LDS)
// Builtin signature (from clang diagnostic): first param is
// 'int __attribute__((vector_size(16))) __device__ *' (AS1 int4*).
typedef int b128_t __attribute__((vector_size(16)));
__device__ __forceinline__ void async_ld_b128(void* lds, const void* gsrc) {
  __builtin_amdgcn_global_load_async_to_lds_b128(
      (__attribute__((address_space(1))) b128_t*)gsrc,
      (__attribute__((address_space(3))) b128_t*)lds,
      /*imm offset=*/0, /*cpol=*/0);
}
#endif

// ---------------------------------------------------------------------------
// Fragment load for V_WMMA_F32_16X16X32_F16.
// A tile in LDS row-major [16][ld] f16 (M x K); B uses the SAME gather when
// the weight is stored row-major [N][K].
// Lanes 0-15 -> row=lane, K={0..7,16..23}; lanes 16-31 -> row=lane-16,
// K={8..15,24..31}: two contiguous 16B segments -> ds_load_b128 pairs.
// ---------------------------------------------------------------------------
__device__ __forceinline__ v16h ld_frag(const _Float16* base, int ld) {
  const int lane = threadIdx.x & 31;
  const _Float16* p = base + (lane & 15) * ld + ((lane >> 4) << 3);
  v8h lo = *(const v8h*)(p);
  v8h hi = *(const v8h*)(p + 16);
  v16h r;
#pragma unroll
  for (int i = 0; i < 8; ++i) { r[i] = lo[i]; r[i + 8] = hi[i]; }
  return r;
}

__device__ __forceinline__ v8f wmma16(v16h a, v16h b, v8f c) {
  return __builtin_amdgcn_wmma_f32_16x16x32_f16(
      false, a, false, b, (short)0, c, false, false);
}

// silu via v_exp + v_rcp (avoid IEEE div expansion)
__device__ __forceinline__ float siluf(float v) {
  return v * __builtin_amdgcn_rcpf(1.f + __expf(-v));
}
__device__ __forceinline__ float softplusf(float v) {
  return (v > 20.f) ? v : __logf(1.f + __expf(v));
}

// ===========================================================================
// Kernel 1: RMSNorm + xz = xn @ W_in^T + b_in ; split into u0 (raw) / z=silu
// ===========================================================================
__global__ void __launch_bounds__(256)
k_norm_in(const float* __restrict__ x, const float* __restrict__ w_norm,
          const float* __restrict__ W_in, const float* __restrict__ b_in,
          float* __restrict__ u0, float* __restrict__ zb) {
  __shared__ __align__(16) _Float16 xnh[16 * DM];        // A: 16x64
  __shared__ __align__(16) _Float16 wlds[2 * DI * DM];   // B: 256x64 (32 KB)
  const int tid = threadIdx.x;
  const int tb  = blockIdx.x * 16;

  for (int i = tid; i < 2 * DI * DM; i += 256) wlds[i] = (_Float16)W_in[i];

  {  // RMSNorm: 16 threads per token, shfl-xor reduce in 16-lane groups
    const int m = tid >> 4, s = tid & 15;
    const float* xr = x + (size_t)(tb + m) * DM + s * 4;
    float p0 = xr[0], p1 = xr[1], p2 = xr[2], p3 = xr[3];
    float ss = p0 * p0 + p1 * p1 + p2 * p2 + p3 * p3;
    ss += __shfl_xor(ss, 1, 16);
    ss += __shfl_xor(ss, 2, 16);
    ss += __shfl_xor(ss, 4, 16);
    ss += __shfl_xor(ss, 8, 16);
    const float sc = rsqrtf(ss * (1.f / DM) + EPSV);
    _Float16* dst = xnh + m * DM + s * 4;
    dst[0] = (_Float16)(p0 * sc * w_norm[s * 4 + 0]);
    dst[1] = (_Float16)(p1 * sc * w_norm[s * 4 + 1]);
    dst[2] = (_Float16)(p2 * sc * w_norm[s * 4 + 2]);
    dst[3] = (_Float16)(p3 * sc * w_norm[s * 4 + 3]);
  }
  __syncthreads();

  const int wv = tid >> 5, lane = tid & 31;
  for (int nt = wv; nt < 16; nt += 8) {
    v8f c = {};
#pragma unroll
    for (int k0 = 0; k0 < DM; k0 += 32) {
      v16h a = ld_frag(xnh + k0, DM);
      v16h b = ld_frag(wlds + nt * 16 * DM + k0, DM);
      c = wmma16(a, b, c);
    }
    const int col  = nt * 16 + (lane & 15);
    const int rb   = (lane >> 4) << 3;
    const float bias = b_in[col];
    if (nt < 8) {          // wave-uniform branch: u half (raw, feeds conv)
#pragma unroll
      for (int v = 0; v < 8; ++v)
        u0[(size_t)(tb + rb + v) * DI + col] = c[v] + bias;
    } else {               // z half: silu gate
#pragma unroll
      for (int v = 0; v < 8; ++v)
        zb[(size_t)(tb + rb + v) * DI + (col - DI)] = siluf(c[v] + bias);
    }
  }
}

// ===========================================================================
// Kernel 2: conv1d (16x512x128 GEMM) + silu -> u ; xp = u@W_xp^T ;
//           delta = softplus(dt @ W_dt^T + b_dt)
// ===========================================================================
__global__ void __launch_bounds__(256)
k_conv_xp(const float* __restrict__ u0, const float* __restrict__ W_conv,
          const float* __restrict__ b_conv, const float* __restrict__ W_xp,
          const float* __restrict__ b_xp, const float* __restrict__ W_dt,
          const float* __restrict__ b_dt,
          float* __restrict__ ub, float* __restrict__ db,
          float* __restrict__ Bb, float* __restrict__ Cb) {
  __shared__ __align__(16) _Float16 wc[DI * 512];   // B: wc[o][k*128+i] (128KB)
  __shared__ __align__(16) _Float16 ac[16 * 512];   // A: ac[m][k*128+i]
  __shared__ __align__(16) _Float16 uh[16 * DI];    // u tile (f16) for xp GEMM
  __shared__ __align__(16) _Float16 wxp[48 * DI];   // W_xp padded 36->48 rows
  __shared__ float xps[16 * 48];
  const int tid = threadIdx.x;
  const int tb  = blockIdx.x * 16;
  const int b   = tb / LL;
  const int l0  = tb % LL;

  for (int idx = tid; idx < DI * 512; idx += 256) {      // rearrange W_conv
    const int o = idx >> 9, kk = idx & 511, k = kk >> 7, i = kk & 127;
    wc[idx] = (_Float16)W_conv[(o * DI + i) * KC + k];
  }
  for (int idx = tid; idx < 48 * DI; idx += 256) {
    const int r = idx >> 7, i = idx & 127;
    wxp[idx] = (r < XPW) ? (_Float16)W_xp[r * DI + i] : (_Float16)0.f;
  }
  for (int idx = tid; idx < 16 * 512; idx += 256) {      // causal A tile
    const int m = idx >> 9, kk = idx & 511, k = kk >> 7, i = kk & 127;
    const int l = l0 + m - (KC - 1) + k;
    const float v = (l >= 0 && l < LL) ? u0[((size_t)b * LL + l) * DI + i] : 0.f;
    ac[idx] = (_Float16)v;
  }
  __syncthreads();

  const int wv = tid >> 5, lane = tid & 31;
  {  // conv GEMM: wave wv owns output-channel tile wv
    v8f c = {};
#pragma unroll
    for (int k0 = 0; k0 < 512; k0 += 32) {
      v16h a  = ld_frag(ac + k0, 512);
      v16h bf = ld_frag(wc + wv * 16 * 512 + k0, 512);
      c = wmma16(a, bf, c);
    }
    const int col = wv * 16 + (lane & 15);
    const int rb  = (lane >> 4) << 3;
    const float bias = b_conv[col];
#pragma unroll
    for (int v = 0; v < 8; ++v) {
      const int row = rb + v;
      const float uu = siluf(c[v] + bias);
      ub[(size_t)(tb + row) * DI + col] = uu;
      uh[row * DI + col] = (_Float16)uu;
    }
  }
  __syncthreads();

  if (wv < 3) {  // xp GEMM: 48 (padded) outputs = 3 N-tiles, K=128
    v8f c = {};
#pragma unroll
    for (int k0 = 0; k0 < DI; k0 += 32) {
      v16h a  = ld_frag(uh + k0, DI);
      v16h bf = ld_frag(wxp + wv * 16 * DI + k0, DI);
      c = wmma16(a, bf, c);
    }
    const int col = wv * 16 + (lane & 15);
    const int rb  = (lane >> 4) << 3;
    const float bias = (col < XPW) ? b_xp[col] : 0.f;
#pragma unroll
    for (int v = 0; v < 8; ++v) {
      const int row = rb + v;
      const float val = c[v] + bias;
      xps[row * 48 + col] = val;
      const size_t t = (size_t)(tb + row);
      if (col >= RK && col < RK + DS)        Bb[t * DS + (col - RK)] = val;
      else if (col >= RK + DS && col < XPW)  Cb[t * DS + (col - RK - DS)] = val;
    }
  }
  __syncthreads();

  for (int idx = tid; idx < 16 * DI; idx += 256) {  // delta (K=4: VALU)
    const int m = idx >> 7, d = idx & 127;
    const float* r = xps + m * 48;
    float acc = b_dt[d];
#pragma unroll
    for (int j = 0; j < RK; ++j) acc = fmaf(r[j], W_dt[d * RK + j], acc);
    db[(size_t)(tb + m) * DI + d] = softplusf(acc);
  }
}

// ===========================================================================
// Kernel 3: selective scan. One block per batch, one thread per channel.
// ===========================================================================
__global__ void __launch_bounds__(128)
k_scan(const float* __restrict__ db, const float* __restrict__ ub,
       const float* __restrict__ zb, const float* __restrict__ Bb,
       const float* __restrict__ Cb, const float* __restrict__ A_log,
       const float* __restrict__ Dp, float* __restrict__ yb) {
  const int d = threadIdx.x;
  const int b = blockIdx.x;
  float Arow[DS];
#pragma unroll
  for (int n = 0; n < DS; ++n) Arow[n] = -__expf(A_log[d * DS + n]);
  const float Dd = Dp[d];
  float h[DS];
#pragma unroll
  for (int n = 0; n < DS; ++n) h[n] = 0.f;

  size_t t = (size_t)b * LL;
  for (int l = 0; l < LL; ++l, ++t) {
    if (l + 1 < LL) {  // hide latency of next step's operands
      __builtin_prefetch((const void*)(db + (t + 1) * DI + d), 0, 1);
      __builtin_prefetch((const void*)(ub + (t + 1) * DI + d), 0, 1);
      __builtin_prefetch((const void*)(Bb + (t + 1) * DS), 0, 1);
    }
    const float delta = db[t * DI + d];
    const float u     = ub[t * DI + d];
    const float z     = zb[t * DI + d];
    const float du    = delta * u;
    const float* Br = Bb + t * DS;
    const float* Cr = Cb + t * DS;
    float y = 0.f;
#pragma unroll
    for (int n = 0; n < DS; ++n) {
      const float a = __expf(delta * Arow[n]);
      h[n] = fmaf(a, h[n], du * Br[n]);
      y    = fmaf(h[n], Cr[n], y);
    }
    yb[t * DI + d] = (y + Dd * u) * z;
  }
}

// ===========================================================================
// Kernel 4: out = x + y @ W_out^T + b_out.
// W_out / y tiles staged via async global->LDS b128 (ASYNCcnt), then f32->f16.
// ===========================================================================
__global__ void __launch_bounds__(128)
k_out(const float* __restrict__ yb, const float* __restrict__ W_out,
      const float* __restrict__ b_out, const float* __restrict__ x,
      float* __restrict__ out) {
  __shared__ __align__(16) float    wstage[DM * DI];  // 32KB raw W_out
  __shared__ __align__(16) float    ystage[16 * DI];  // 8KB raw y tile
  __shared__ __align__(16) _Float16 wl[DM * DI];
  __shared__ __align__(16) _Float16 yl[16 * DI];
  const int tid = threadIdx.x;
  const int tb  = blockIdx.x * 16;

#if defined(USE_ASYNC_LDS)
  for (int i = tid * 4; i < DM * DI; i += 128 * 4)
    async_ld_b128(&wstage[i], &W_out[i]);
  for (int i = tid * 4; i < 16 * DI; i += 128 * 4)
    async_ld_b128(&ystage[i], &yb[(size_t)tb * DI + i]);
  __builtin_amdgcn_s_wait_asynccnt(0);
#else
  for (int i = tid; i < DM * DI; i += 128) wstage[i] = W_out[i];
  for (int i = tid; i < 16 * DI; i += 128) ystage[i] = yb[(size_t)tb * DI + i];
#endif
  __syncthreads();
  for (int i = tid; i < DM * DI; i += 128) wl[i] = (_Float16)wstage[i];
  for (int i = tid; i < 16 * DI; i += 128) yl[i] = (_Float16)ystage[i];
  __syncthreads();

  const int wv = tid >> 5, lane = tid & 31;
  v8f c = {};
#pragma unroll
  for (int k0 = 0; k0 < DI; k0 += 32) {
    v16h a  = ld_frag(yl + k0, DI);
    v16h bf = ld_frag(wl + wv * 16 * DI + k0, DI);
    c = wmma16(a, bf, c);
  }
  const int col = wv * 16 + (lane & 15);
  const int rb  = (lane >> 4) << 3;
  const float bias = b_out[col];
#pragma unroll
  for (int v = 0; v < 8; ++v) {
    const size_t t = (size_t)(tb + rb + v);
    out[t * DM + col] = x[t * DM + col] + c[v] + bias;
  }
}

// ===========================================================================
extern "C" void kernel_launch(void* const* d_in, const int* in_sizes, int n_in,
                              void* d_out, int out_size, void* d_ws, size_t ws_size,
                              hipStream_t stream) {
  const float* x      = (const float*)d_in[0];
  const float* w_norm = (const float*)d_in[1];
  const float* W_in   = (const float*)d_in[2];
  const float* b_in   = (const float*)d_in[3];
  const float* W_conv = (const float*)d_in[4];
  const float* b_conv = (const float*)d_in[5];
  const float* W_xp   = (const float*)d_in[6];
  const float* b_xp   = (const float*)d_in[7];
  const float* W_dt   = (const float*)d_in[8];
  const float* b_dt   = (const float*)d_in[9];
  const float* W_out  = (const float*)d_in[10];
  const float* b_out  = (const float*)d_in[11];
  const float* A_log  = (const float*)d_in[12];
  const float* Dp     = (const float*)d_in[13];
  float* out = (float*)d_out;

  float* ws = (float*)d_ws;        // fp32 planes, all fully written before read
  float* u0 = ws;                                  // TOK*DI (conv input)
  float* zb = u0 + (size_t)TOK * DI;               // TOK*DI (silu gate)
  float* ub = zb + (size_t)TOK * DI;               // TOK*DI (post-conv silu)
  float* db = ub + (size_t)TOK * DI;               // TOK*DI (delta)
  float* Bb = db + (size_t)TOK * DI;               // TOK*DS
  float* Cb = Bb + (size_t)TOK * DS;               // TOK*DS
  float* yb = Cb + (size_t)TOK * DS;               // TOK*DI (gated scan out)

  k_norm_in<<<NTIL, 256, 0, stream>>>(x, w_norm, W_in, b_in, u0, zb);
  k_conv_xp<<<NTIL, 256, 0, stream>>>(u0, W_conv, b_conv, W_xp, b_xp, W_dt, b_dt,
                                      ub, db, Bb, Cb);
  k_scan<<<BB, 128, 0, stream>>>(db, ub, zb, Bb, Cb, A_log, Dp, yb);
  k_out<<<NTIL, 128, 0, stream>>>(yb, W_out, b_out, x, out);
}
